// GINWithEdgeFeatures_53283364274718
// MI455X (gfx1250) — matrix operand order repaired and checked by reading, not compiled, not run on hardware
//
#include <hip/hip_runtime.h>
#include <hip/hip_bf16.h>
#include <stdint.h>

// ---------------- problem constants ----------------
#define E_EDGES   300000
#define NN_NODES  50000
#define IN_CH     128
#define HID       256
#define N_GRAPHS  512
#define OUT_CH    128
#define BN_EPS    1e-5f

// ---------------- WMMA types ----------------
typedef __bf16 bf16_t;
typedef bf16_t v16bf __attribute__((ext_vector_type(16)));
typedef float  v8f   __attribute__((ext_vector_type(8)));
typedef int    v4i   __attribute__((ext_vector_type(4)));

union Frag {
  v16bf v;
  uint32_t u[8];
  uint4 q[2];
};

#if __has_builtin(__builtin_amdgcn_global_load_async_to_lds_b128) && \
    __has_builtin(__builtin_amdgcn_s_wait_asynccnt)
#define USE_ASYNC_LDS 1
typedef __attribute__((address_space(1))) v4i* gvec_p;
typedef __attribute__((address_space(3))) v4i* lvec_p;
#endif

static __device__ __forceinline__ uint16_t f32_to_bf16(float f) {
  bf16_t h = (bf16_t)f;   // hardware cvt, RNE
  return __builtin_bit_cast(uint16_t, h);
}

// A-matrix (16x32 bf16) per-lane K offset for VGPR r, lane-half h:
// lanes 0-15:  r0..3 -> K{0,1..6,7},  r4..7 -> K{16,17..22,23}
// lanes 16-31: r0..3 -> K{8,9..14,15},r4..7 -> K{24,25..30,31}
static __device__ __forceinline__ int a_koff(int r, int half) {
  return (r < 4 ? 2 * r : 16 + 2 * (r - 4)) + 8 * half;
}

static __device__ __forceinline__ v8f wmma_bf16(const Frag& a, const Frag& b, v8f c) {
  return __builtin_amdgcn_wmma_f32_16x16x32_bf16(false, a.v, false, b.v, (short)0, c,
                                                 false, false);
}

static __device__ __forceinline__ v8f vzero8() {
  v8f z = {0.f, 0.f, 0.f, 0.f, 0.f, 0.f, 0.f, 0.f};
  return z;
}

// ---------------- weight packing ----------------
// B fragment layout (32x16 bf16): lane L: n = L&15; half = L>>4;
// VGPR r holds K = (r<4 ? 2r : 16+2(r-4)) + 8*half, +1.
// Packed as uint32 pairs: idx = ((kt*ntiles + ct)*32 + lane)*8 + r
__global__ void pack_w_kernel(const float* __restrict__ W, uint32_t* __restrict__ out,
                              int Ksrc, int Kpad, int N, int total) {
  int idx = blockIdx.x * 256 + threadIdx.x;
  if (idx >= total) return;
  int r    = idx & 7;
  int lane = (idx >> 3) & 31;
  int tile = idx >> 8;
  int ntiles = N >> 4;
  int kt = tile / ntiles, ct = tile - kt * ntiles;
  int half = lane >> 4;
  int n = ct * 16 + (lane & 15);
  int k = kt * 32 + (r < 4 ? 2 * r : 16 + 2 * (r - 4)) + 8 * half;
  uint16_t lo = (k     < Ksrc) ? f32_to_bf16(W[(size_t)k * N + n])       : (uint16_t)0;
  uint16_t hi = (k + 1 < Ksrc) ? f32_to_bf16(W[(size_t)(k + 1) * N + n]) : (uint16_t)0;
  out[idx] = (uint32_t)lo | ((uint32_t)hi << 16);
  (void)Kpad;
}

// ------- fused edge MLP: Lin(16->256)+ReLU -> Lin(256->256), 64 edges per block -------
__global__ void edge_mlp_kernel(const float* __restrict__ ea,
                                const uint32_t* __restrict__ W1p, const float* __restrict__ b1,
                                const uint32_t* __restrict__ W2p, const float* __restrict__ b2,
                                uint16_t* __restrict__ edge_emb) {
  const int e0   = blockIdx.x * 64;
  const int lane = threadIdx.x & 31;
  const int wave = threadIdx.x >> 5;
  __shared__ uint16_t sA0[64 * 40];    // 64 edges x 32 (K padded 16->32), stride 40
  __shared__ uint16_t sHid[64 * 264];  // 64 edges x 256 hidden, stride 264
  {
    int t = threadIdx.x;
    int row = t >> 2, c4 = (t & 3) * 4;
    int e = e0 + row;
#pragma unroll
    for (int j = 0; j < 4; ++j) {
      float v = (e < E_EDGES) ? ea[(size_t)e * 16 + c4 + j] : 0.f;
      sA0[row * 40 + c4 + j]      = f32_to_bf16(v);
      sA0[row * 40 + 16 + c4 + j] = 0;
    }
  }
  __syncthreads();
  const int m = lane & 15, half = lane >> 4;
  // GEMM1: [64 x 32] @ [32 x 256] -> ReLU -> sHid
  for (int ct = wave; ct < 16; ct += 8) {
    Frag b;
    const uint32_t* bp = W1p + ((size_t)ct * 32 + lane) * 8;
    b.q[0] = *(const uint4*)bp;
    b.q[1] = *(const uint4*)(bp + 4);
    int n = ct * 16 + m;
    float bias = b1[n];
#pragma unroll
    for (int rt = 0; rt < 4; ++rt) {
      Frag a;
#pragma unroll
      for (int r = 0; r < 8; ++r)
        a.u[r] = *(const uint32_t*)&sA0[(rt * 16 + m) * 40 + a_koff(r, half)];
      v8f acc = vzero8();
      acc = wmma_bf16(a, b, acc);
#pragma unroll
      for (int r = 0; r < 8; ++r) {
        int row = rt * 16 + r + half * 8;
        float v = fmaxf(acc[r] + bias, 0.f);
        sHid[row * 264 + n] = f32_to_bf16(v);
      }
    }
  }
  __syncthreads();
  // GEMM2: [64 x 256] @ [256 x 256], kt-outer, 4 persistent accumulators
  for (int ct = wave; ct < 16; ct += 8) {
    v8f acc[4];
#pragma unroll
    for (int rt = 0; rt < 4; ++rt) acc[rt] = vzero8();
    for (int kt = 0; kt < 8; ++kt) {
      Frag b;
      const uint32_t* bp = W2p + ((size_t)(kt * 16 + ct) * 32 + lane) * 8;
      b.q[0] = *(const uint4*)bp;
      b.q[1] = *(const uint4*)(bp + 4);
#pragma unroll
      for (int rt = 0; rt < 4; ++rt) {
        Frag a;
#pragma unroll
        for (int r = 0; r < 8; ++r)
          a.u[r] = *(const uint32_t*)&sHid[(rt * 16 + m) * 264 + kt * 32 + a_koff(r, half)];
        acc[rt] = wmma_bf16(a, b, acc[rt]);
      }
    }
    int n = ct * 16 + m;
    float bias = b2[n];
#pragma unroll
    for (int rt = 0; rt < 4; ++rt) {
#pragma unroll
      for (int r = 0; r < 8; ++r) {
        int row = rt * 16 + r + half * 8;
        int e = e0 + row;
        if (e < E_EDGES)
          edge_emb[(size_t)e * HID + n] = f32_to_bf16(acc[rt][r] + bias);
      }
    }
  }
}

// ------- fused GINE message, 64 edges per block:
// e = edge_emb@We + be; msg = relu(h[src]+e); agg[dst] += msg -------
__global__ void msg_kernel(const uint16_t* __restrict__ edge_emb,
                           const uint32_t* __restrict__ Wp, const float* __restrict__ bias,
                           const float* __restrict__ hsrc,
                           const int* __restrict__ src, const int* __restrict__ dst,
                           float* __restrict__ agg, int Nd) {
  const int e0   = blockIdx.x * 64;
  const int lane = threadIdx.x & 31;
  const int wave = threadIdx.x >> 5;
  __shared__ uint16_t sA[64 * 264];
  __shared__ int sSrc[64], sDst[64];
  {
    int t = threadIdx.x;
    int row = t >> 2;                 // 0..63, 4 threads per row
    int e = e0 + row;
    if (e < E_EDGES) {
      const uint16_t* g = edge_emb + (size_t)e * HID;
#pragma unroll
      for (int j = 0; j < 8; ++j) {
        int q = (t & 3) + 4 * j;      // uint4 index within the 256-elem row
#ifdef USE_ASYNC_LDS
        __builtin_amdgcn_global_load_async_to_lds_b128(
            (gvec_p)(g + q * 8), (lvec_p)&sA[row * 264 + q * 8], 0, 0);
#else
        *(uint4*)&sA[row * 264 + q * 8] = *(const uint4*)(g + q * 8);
#endif
      }
    } else {
      uint4 z = {0u, 0u, 0u, 0u};
#pragma unroll
      for (int j = 0; j < 8; ++j) {
        int q = (t & 3) + 4 * j;
        *(uint4*)&sA[row * 264 + q * 8] = z;
      }
    }
    if (t < 64) {
      int e2 = e0 + t;
      sSrc[t] = (e2 < E_EDGES) ? src[e2] : 0;
      sDst[t] = (e2 < E_EDGES) ? dst[e2] : 0;
    }
  }
#ifdef USE_ASYNC_LDS
  __builtin_amdgcn_s_wait_asynccnt(0);
#endif
  __syncthreads();
  // prefetch the gather rows (random-access, latency-exposed)
  {
    int t = threadIdx.x;
    int row = t >> 2;
    int seg = (t & 3) * (Nd >> 2);
    __builtin_prefetch(&hsrc[(size_t)sSrc[row] * Nd + seg], 0, 0);
  }
  const int m = lane & 15, half = lane >> 4;
  const int ntiles = Nd >> 4;
  for (int ct = wave; ct < ntiles; ct += 8) {
    v8f acc[4];
#pragma unroll
    for (int rt = 0; rt < 4; ++rt) acc[rt] = vzero8();
    for (int kt = 0; kt < 8; ++kt) {  // K = HID = 256
      Frag b;
      const uint32_t* bp = Wp + ((size_t)(kt * ntiles + ct) * 32 + lane) * 8;
      b.q[0] = *(const uint4*)bp;
      b.q[1] = *(const uint4*)(bp + 4);
#pragma unroll
      for (int rt = 0; rt < 4; ++rt) {
        Frag a;
#pragma unroll
        for (int r = 0; r < 8; ++r)
          a.u[r] = *(const uint32_t*)&sA[(rt * 16 + m) * 264 + kt * 32 + a_koff(r, half)];
        acc[rt] = wmma_bf16(a, b, acc[rt]);
      }
    }
    int n = ct * 16 + m;
    float bb = bias[n];
#pragma unroll
    for (int rt = 0; rt < 4; ++rt) {
#pragma unroll
      for (int r = 0; r < 8; ++r) {
        int row = rt * 16 + r + half * 8;
        int e = e0 + row;
        if (e < E_EDGES) {
          float msg = acc[rt][r] + bb + hsrc[(size_t)sSrc[row] * Nd + n];
          msg = fmaxf(msg, 0.f);
          atomicAdd(&agg[(size_t)sDst[row] * Nd + n], msg);
        }
      }
    }
  }
}

// ------- wide GEMM (64 rows, K<=256), A = fp32 (A1 [+ A2]); optional BN; out bf16 -------
__global__ void gemm_f32A_k64(const float* __restrict__ A1, const float* __restrict__ A2,
                              int M, int K, int Nout,
                              const uint32_t* __restrict__ Wp, const float* __restrict__ bias,
                              const float* __restrict__ gamma, const float* __restrict__ beta,
                              const float* __restrict__ mean, const float* __restrict__ var,
                              int relu, uint16_t* __restrict__ outb) {
  const int m0   = blockIdx.x * 64;
  const int lane = threadIdx.x & 31;
  const int wave = threadIdx.x >> 5;
  const int Kp = K + 8;
  __shared__ uint16_t sA[64 * 264];    // max K = 256
  {
    int t = threadIdx.x;
    int row = t >> 2;
    int rg = m0 + row;
    if (rg < M) {
      size_t gbase = (size_t)rg * K;
      const float4* a1 = (const float4*)(A1 + gbase);
      const float4* a2 = A2 ? (const float4*)(A2 + gbase) : nullptr;
      for (int c4 = (t & 3); c4 < (K >> 2); c4 += 4) {
        float4 v = a1[c4];
        if (a2) { float4 w = a2[c4]; v.x += w.x; v.y += w.y; v.z += w.z; v.w += w.w; }
        uint16_t* s = &sA[row * Kp + c4 * 4];
        s[0] = f32_to_bf16(v.x); s[1] = f32_to_bf16(v.y);
        s[2] = f32_to_bf16(v.z); s[3] = f32_to_bf16(v.w);
      }
    } else {
      for (int c4 = (t & 3); c4 < (K >> 2); c4 += 4) {
        uint16_t* s = &sA[row * Kp + c4 * 4];
        s[0] = 0; s[1] = 0; s[2] = 0; s[3] = 0;
      }
    }
  }
  __syncthreads();
  const int m = lane & 15, half = lane >> 4;
  const int ntiles = Nout >> 4, ktiles = K >> 5;
  for (int ct = wave; ct < ntiles; ct += 8) {
    v8f acc[4];
#pragma unroll
    for (int rt = 0; rt < 4; ++rt) acc[rt] = vzero8();
    for (int kt = 0; kt < ktiles; ++kt) {
      Frag b;
      const uint32_t* bp = Wp + ((size_t)(kt * ntiles + ct) * 32 + lane) * 8;
      b.q[0] = *(const uint4*)bp;
      b.q[1] = *(const uint4*)(bp + 4);
#pragma unroll
      for (int rt = 0; rt < 4; ++rt) {
        Frag a;
#pragma unroll
        for (int r = 0; r < 8; ++r)
          a.u[r] = *(const uint32_t*)&sA[(rt * 16 + m) * Kp + kt * 32 + a_koff(r, half)];
        acc[rt] = wmma_bf16(a, b, acc[rt]);
      }
    }
    int n = ct * 16 + m;
    float bb = bias[n];
    float g = 1.f, be = 0.f, mu = 0.f, inv = 1.f;
    if (gamma) {
      g = gamma[n]; be = beta[n]; mu = mean[n]; inv = rsqrtf(var[n] + BN_EPS);
    }
#pragma unroll
    for (int rt = 0; rt < 4; ++rt) {
#pragma unroll
      for (int r = 0; r < 8; ++r) {
        int rg = m0 + rt * 16 + r + half * 8;
        if (rg < M) {
          float v = acc[rt][r] + bb;
          if (gamma) v = g * (v - mu) * inv + be;
          if (relu) v = fmaxf(v, 0.f);
          outb[(size_t)rg * Nout + n] = f32_to_bf16(v);
        }
      }
    }
  }
}

// ------- wide GEMM (64 rows, K<=256), A = bf16; out fp32 + ReLU + pool atomics -------
__global__ void gemm_bf16A_k64(const uint16_t* __restrict__ A, int M, int K, int Nout,
                               const uint32_t* __restrict__ Wp, const float* __restrict__ bias,
                               int relu, float* __restrict__ outf,
                               const int* __restrict__ batch, float* __restrict__ pool,
                               int poolOff) {
  const int m0   = blockIdx.x * 64;
  const int lane = threadIdx.x & 31;
  const int wave = threadIdx.x >> 5;
  const int Kp = K + 8;
  __shared__ uint16_t sA[64 * 264];
  __shared__ int sB[64];
  {
    int t = threadIdx.x;
    int row = t >> 2;
    int rg = m0 + row;
    if (rg < M) {
      const uint16_t* g = A + (size_t)rg * K;
      for (int c8 = (t & 3); c8 < (K >> 3); c8 += 4)
        *(uint4*)&sA[row * Kp + c8 * 8] = *(const uint4*)(g + c8 * 8);
    } else {
      uint4 z = {0u, 0u, 0u, 0u};
      for (int c8 = (t & 3); c8 < (K >> 3); c8 += 4)
        *(uint4*)&sA[row * Kp + c8 * 8] = z;
    }
    if (t < 64) {
      int rg2 = m0 + t;
      sB[t] = (batch && rg2 < M) ? batch[rg2] : 0;
    }
  }
  __syncthreads();
  const int m = lane & 15, half = lane >> 4;
  const int ntiles = Nout >> 4, ktiles = K >> 5;
  for (int ct = wave; ct < ntiles; ct += 8) {
    v8f acc[4];
#pragma unroll
    for (int rt = 0; rt < 4; ++rt) acc[rt] = vzero8();
    for (int kt = 0; kt < ktiles; ++kt) {
      Frag b;
      const uint32_t* bp = Wp + ((size_t)(kt * ntiles + ct) * 32 + lane) * 8;
      b.q[0] = *(const uint4*)bp;
      b.q[1] = *(const uint4*)(bp + 4);
#pragma unroll
      for (int rt = 0; rt < 4; ++rt) {
        Frag a;
#pragma unroll
        for (int r = 0; r < 8; ++r)
          a.u[r] = *(const uint32_t*)&sA[(rt * 16 + m) * Kp + kt * 32 + a_koff(r, half)];
        acc[rt] = wmma_bf16(a, b, acc[rt]);
      }
    }
    int n = ct * 16 + m;
    float bb = bias[n];
#pragma unroll
    for (int rt = 0; rt < 4; ++rt) {
#pragma unroll
      for (int r = 0; r < 8; ++r) {
        int row = rt * 16 + r + half * 8;
        int rg = m0 + row;
        if (rg < M) {
          float v = acc[rt][r] + bb;
          if (relu) v = fmaxf(v, 0.f);
          outf[(size_t)rg * Nout + n] = v;
          if (batch) atomicAdd(&pool[(size_t)sB[row] * 768 + poolOff + n], v);
        }
      }
    }
  }
}

// ------- narrow GEMM (16 rows, K<=768) for the FC layers, A = fp32 -> out bf16 -------
__global__ void gemm_f32A_k16(const float* __restrict__ A1, int K, int Nout,
                              const uint32_t* __restrict__ Wp, const float* __restrict__ bias,
                              int relu, uint16_t* __restrict__ outb) {
  const int m0   = blockIdx.x * 16;
  const int lane = threadIdx.x & 31;
  const int wave = threadIdx.x >> 5;
  const int Kp = K + 8;
  __shared__ uint16_t sA[16 * (768 + 8)];
  {
    int row = threadIdx.x >> 4;
    size_t gbase = (size_t)(m0 + row) * K;
    for (int c = (threadIdx.x & 15); c < K; c += 16)
      sA[row * Kp + c] = f32_to_bf16(A1[gbase + c]);
  }
  __syncthreads();
  const int m = lane & 15, half = lane >> 4;
  const int ntiles = Nout >> 4, ktiles = K >> 5;
  for (int ct = wave; ct < ntiles; ct += 8) {
    v8f acc = vzero8();
    for (int kt = 0; kt < ktiles; ++kt) {
      Frag a, b;
#pragma unroll
      for (int r = 0; r < 8; ++r)
        a.u[r] = *(const uint32_t*)&sA[m * Kp + kt * 32 + a_koff(r, half)];
      const uint32_t* bp = Wp + ((size_t)(kt * ntiles + ct) * 32 + lane) * 8;
      b.q[0] = *(const uint4*)bp;
      b.q[1] = *(const uint4*)(bp + 4);
      acc = wmma_bf16(a, b, acc);
    }
    int n = ct * 16 + m;
    float bb = bias[n];
#pragma unroll
    for (int r = 0; r < 8; ++r) {
      float v = acc[r] + bb;
      if (relu) v = fmaxf(v, 0.f);
      outb[(size_t)(m0 + r + half * 8) * Nout + n] = f32_to_bf16(v);
    }
  }
}

// ------- narrow GEMM (16 rows, K<=768) for the FC layers, A = bf16 -> out fp32 -------
__global__ void gemm_bf16A_k16(const uint16_t* __restrict__ A, int K, int Nout,
                               const uint32_t* __restrict__ Wp, const float* __restrict__ bias,
                               int relu, float* __restrict__ outf) {
  const int m0   = blockIdx.x * 16;
  const int lane = threadIdx.x & 31;
  const int wave = threadIdx.x >> 5;
  const int Kp = K + 8;
  __shared__ uint16_t sA[16 * (768 + 8)];
  {
    int row = threadIdx.x >> 4;
    size_t gbase = (size_t)(m0 + row) * K;
    for (int c8 = (threadIdx.x & 15); c8 < (K >> 3); c8 += 16)
      *(uint4*)&sA[row * Kp + c8 * 8] = *(const uint4*)(A + gbase + (size_t)c8 * 8);
  }
  __syncthreads();
  const int m = lane & 15, half = lane >> 4;
  const int ntiles = Nout >> 4, ktiles = K >> 5;
  for (int ct = wave; ct < ntiles; ct += 8) {
    v8f acc = vzero8();
    for (int kt = 0; kt < ktiles; ++kt) {
      Frag a, b;
#pragma unroll
      for (int r = 0; r < 8; ++r)
        a.u[r] = *(const uint32_t*)&sA[m * Kp + kt * 32 + a_koff(r, half)];
      const uint32_t* bp = Wp + ((size_t)(kt * ntiles + ct) * 32 + lane) * 8;
      b.q[0] = *(const uint4*)bp;
      b.q[1] = *(const uint4*)(bp + 4);
      acc = wmma_bf16(a, b, acc);
    }
    int n = ct * 16 + m;
    float bb = bias[n];
#pragma unroll
    for (int r = 0; r < 8; ++r) {
      float v = acc[r] + bb;
      if (relu) v = fmaxf(v, 0.f);
      outf[(size_t)(m0 + r + half * 8) * Nout + n] = v;
    }
  }
}

// ---------------- host launch ----------------
extern "C" void kernel_launch(void* const* d_in, const int* in_sizes, int n_in,
                              void* d_out, int out_size, void* d_ws, size_t ws_size,
                              hipStream_t stream) {
  (void)in_sizes; (void)n_in; (void)out_size; (void)ws_size;
  const float* x    = (const float*)d_in[0];
  const int*   eidx = (const int*)d_in[1];
  const float* ea   = (const float*)d_in[2];
  const int*   batch= (const int*)d_in[3];
  const float* eW1  = (const float*)d_in[4];  const float* eb1  = (const float*)d_in[5];
  const float* eW2  = (const float*)d_in[6];  const float* eb2  = (const float*)d_in[7];
  const float* m1W1 = (const float*)d_in[8];  const float* m1b1 = (const float*)d_in[9];
  const float* m1W2 = (const float*)d_in[10]; const float* m1b2 = (const float*)d_in[11];
  const float* m1g  = (const float*)d_in[12]; const float* m1be = (const float*)d_in[13];
  const float* m1mu = (const float*)d_in[14]; const float* m1va = (const float*)d_in[15];
  const float* mhW1 = (const float*)d_in[16]; const float* mhb1 = (const float*)d_in[17];
  const float* mhW2 = (const float*)d_in[18]; const float* mhb2 = (const float*)d_in[19];
  const float* mhg  = (const float*)d_in[20]; const float* mhbe = (const float*)d_in[21];
  const float* mhmu = (const float*)d_in[22]; const float* mhva = (const float*)d_in[23];
  const float* le1W = (const float*)d_in[24]; const float* le1b = (const float*)d_in[25];
  const float* le2W = (const float*)d_in[26]; const float* le2b = (const float*)d_in[27];
  const float* le3W = (const float*)d_in[28]; const float* le3b = (const float*)d_in[29];
  const float* f1W  = (const float*)d_in[30]; const float* f1b  = (const float*)d_in[31];
  const float* f2W  = (const float*)d_in[32]; const float* f2b  = (const float*)d_in[33];
  const int* src = eidx;
  const int* dst = eidx + E_EDGES;

  uint8_t* ws = (uint8_t*)d_ws;
  size_t off = 0;
  auto take = [&](size_t bytes) -> void* {
    void* p = ws + off;
    off = (off + bytes + 255) & ~(size_t)255;
    return p;
  };
  uint16_t* edge_emb = (uint16_t*)take((size_t)E_EDGES * HID * 2);
  float*    agg      = (float*)take((size_t)NN_NODES * HID * 4);
  float*    hA       = (float*)take((size_t)NN_NODES * HID * 4);
  float*    hB       = (float*)take((size_t)NN_NODES * HID * 4);
  uint16_t* tbuf     = (uint16_t*)take((size_t)NN_NODES * HID * 2);
  float*    pool     = (float*)take((size_t)N_GRAPHS * 3 * HID * 4);
  uint32_t* eW1p  = (uint32_t*)take((size_t)32  * 256 * 2);
  uint32_t* eW2p  = (uint32_t*)take((size_t)256 * 256 * 2);
  uint32_t* m1W1p = (uint32_t*)take((size_t)128 * 256 * 2);
  uint32_t* m1W2p = (uint32_t*)take((size_t)256 * 256 * 2);
  uint32_t* mhW1p = (uint32_t*)take((size_t)256 * 256 * 2);
  uint32_t* mhW2p = (uint32_t*)take((size_t)256 * 256 * 2);
  uint32_t* le1p  = (uint32_t*)take((size_t)256 * 128 * 2);
  uint32_t* le2p  = (uint32_t*)take((size_t)256 * 256 * 2);
  uint32_t* le3p  = (uint32_t*)take((size_t)256 * 256 * 2);
  uint32_t* f1p   = (uint32_t*)take((size_t)768 * 768 * 2);
  uint32_t* f2p   = (uint32_t*)take((size_t)768 * 128 * 2);

  auto pack = [&](const float* W, uint32_t* out, int Ksrc, int Kpad, int N) {
    int total = (Kpad / 32) * (N / 16) * 256;
    pack_w_kernel<<<(total + 255) / 256, 256, 0, stream>>>(W, out, Ksrc, Kpad, N, total);
  };
  pack(eW1,  eW1p,  16,  32,  256);
  pack(eW2,  eW2p,  256, 256, 256);
  pack(m1W1, m1W1p, 128, 128, 256);
  pack(m1W2, m1W2p, 256, 256, 256);
  pack(mhW1, mhW1p, 256, 256, 256);
  pack(mhW2, mhW2p, 256, 256, 256);
  pack(le1W, le1p,  256, 256, 128);
  pack(le2W, le2p,  256, 256, 256);
  pack(le3W, le3p,  256, 256, 256);
  pack(f1W,  f1p,   768, 768, 768);
  pack(f2W,  f2p,   768, 768, 128);

  (void)hipMemsetAsync(pool, 0, (size_t)N_GRAPHS * 3 * HID * 4, stream);

  const int egrid = (E_EDGES + 63) / 64;
  const int ngrid = (NN_NODES + 63) / 64;
  edge_mlp_kernel<<<egrid, 256, 0, stream>>>(ea, eW1p, eb1, eW2p, eb2, edge_emb);

  // ---- conv1 (node dim 128) ----
  (void)hipMemsetAsync(agg, 0, (size_t)NN_NODES * HID * 4, stream);
  msg_kernel<<<egrid, 256, 0, stream>>>(edge_emb, le1p, le1b, x, src, dst, agg, IN_CH);
  gemm_f32A_k64<<<ngrid, 256, 0, stream>>>(agg, x, NN_NODES, IN_CH, HID, m1W1p, m1b1,
                                           m1g, m1be, m1mu, m1va, 1, tbuf);
  gemm_bf16A_k64<<<ngrid, 256, 0, stream>>>(tbuf, NN_NODES, HID, HID, m1W2p, m1b2, 1, hA,
                                            batch, pool, 0);
  // ---- conv2 ----
  (void)hipMemsetAsync(agg, 0, (size_t)NN_NODES * HID * 4, stream);
  msg_kernel<<<egrid, 256, 0, stream>>>(edge_emb, le2p, le2b, hA, src, dst, agg, HID);
  gemm_f32A_k64<<<ngrid, 256, 0, stream>>>(agg, hA, NN_NODES, HID, HID, mhW1p, mhb1,
                                           mhg, mhbe, mhmu, mhva, 1, tbuf);
  gemm_bf16A_k64<<<ngrid, 256, 0, stream>>>(tbuf, NN_NODES, HID, HID, mhW2p, mhb2, 1, hB,
                                            batch, pool, HID);
  // ---- conv3 ----
  (void)hipMemsetAsync(agg, 0, (size_t)NN_NODES * HID * 4, stream);
  msg_kernel<<<egrid, 256, 0, stream>>>(edge_emb, le3p, le3b, hB, src, dst, agg, HID);
  gemm_f32A_k64<<<ngrid, 256, 0, stream>>>(agg, hB, NN_NODES, HID, HID, mhW1p, mhb1,
                                           mhg, mhbe, mhmu, mhva, 1, tbuf);
  gemm_bf16A_k64<<<ngrid, 256, 0, stream>>>(tbuf, NN_NODES, HID, HID, mhW2p, mhb2, 1, hA,
                                            batch, pool, 2 * HID);
  // ---- readout MLP ----
  gemm_f32A_k16<<<N_GRAPHS / 16, 256, 0, stream>>>(pool, 3 * HID, 3 * HID, f1p, f1b, 1, tbuf);
  gemm_bf16A_k16<<<N_GRAPHS / 16, 256, 0, stream>>>(tbuf, 3 * HID, OUT_CH, f2p, f2b, 0,
                                                    (float*)d_out);
}
